// VectorQuantizer_56049323213329
// MI455X (gfx1250) — matrix operand order repaired and checked by reading, not compiled, not run on hardware
//
#include <hip/hip_runtime.h>

// Problem constants (from reference): x (16,64,64,64) f32, embed (64,1024) f32
#define NB   16
#define NC   64
#define NHW  4096
#define NN   65536      // NB*NHW rows
#define NK   1024       // codes
#define ROWS_PER_BLK 128

typedef _Float16 v16h __attribute__((ext_vector_type(16)));
typedef _Float16 v8h  __attribute__((ext_vector_type(8)));
typedef float    v8f  __attribute__((ext_vector_type(8)));
typedef int      v4i  __attribute__((ext_vector_type(4)));

// ---------------------------------------------------------------------------
// gfx1250 async global->LDS copy (ASYNCcnt-tracked DMA), with safe fallback.
// Builtin signature (from hipcc diagnostic): params are
//   (int4 addrspace(1)*, int4 addrspace(3)*, imm offset, imm cpol)
// ---------------------------------------------------------------------------
#if defined(__gfx1250__) && \
    __has_builtin(__builtin_amdgcn_global_load_async_to_lds_b128) && \
    __has_builtin(__builtin_amdgcn_s_wait_asynccnt)
#define VQ_ASYNC_LDS 1
#else
#define VQ_ASYNC_LDS 0
#endif

typedef __attribute__((address_space(1))) v4i* vq_gptr_t;
typedef __attribute__((address_space(3))) v4i* vq_lptr_t;

__device__ __forceinline__ void vq_async_cp16(const void* g, void* l) {
#if VQ_ASYNC_LDS
  __builtin_amdgcn_global_load_async_to_lds_b128((vq_gptr_t)g, (vq_lptr_t)l,
                                                 0, 0);
#else
  *(uint4*)l = *(const uint4*)g;
#endif
}

__device__ __forceinline__ void vq_async_wait0() {
#if VQ_ASYNC_LDS
  __builtin_amdgcn_s_wait_asynccnt(0);
#endif
}

// ---------------------------------------------------------------------------
// Prep: embed f32 -> f16 workspace copy TRANSPOSED to [k][c] (so the per-block
// LDS staging is a pure contiguous DMA), per-code squared norms, zero loss.
// ---------------------------------------------------------------------------
__global__ void vq_prep(const float* __restrict__ embed,
                        _Float16* __restrict__ wsE,
                        float* __restrict__ wsEnorm,
                        float* __restrict__ lossPtr) {
  int gid = blockIdx.x * blockDim.x + threadIdx.x;   // 64 blocks * 256 = 16384
  int i0 = gid * 4;                                  // 4 consecutive k, same c
  if (i0 < NC * NK) {
    int c = i0 >> 10, k0 = i0 & (NK - 1);
    float4 v = *(const float4*)(embed + i0);
    wsE[(k0 + 0) * NC + c] = (_Float16)v.x;
    wsE[(k0 + 1) * NC + c] = (_Float16)v.y;
    wsE[(k0 + 2) * NC + c] = (_Float16)v.z;
    wsE[(k0 + 3) * NC + c] = (_Float16)v.w;
  }
  if (gid < NK) {
    float s = 0.f;
    #pragma unroll 8
    for (int c = 0; c < NC; ++c) {
      float e = embed[c * NK + gid];
      s += e * e;
    }
    wsEnorm[gid] = s;
  }
  if (gid == 0) *lossPtr = 0.f;
}

// ---------------------------------------------------------------------------
// Main: fused distance-GEMM (WMMA f16->f32) + argmin + gather + loss
// LDS: full f16 codebook ([k][c], 128KB, async-DMA'd) + enorm (4KB)
//      + x tile [r][c] f16 (16KB) + per-row argmin (512B)  ~= 152KB of 320KB
// ---------------------------------------------------------------------------
__global__ void __launch_bounds__(256)
vq_main(const float* __restrict__ x, const float* __restrict__ embed,
        const _Float16* __restrict__ wsE, const float* __restrict__ wsEnorm,
        float* __restrict__ outQ, int* __restrict__ outIdx,
        float* __restrict__ lossPtr) {
  extern __shared__ char smem[];
  _Float16* ldsE = (_Float16*)(smem);              // [k][c]  1024*64 f16 = 131072 B
  float*    enS  = (float*)(smem + 131072);        // 1024 f32           =   4096 B
  _Float16* ldsX = (_Float16*)(smem + 135168);     // [r][c]  128*64 f16 =  16384 B
  int*      ldsI = (int*)(smem + 151552);          // 128 int            =    512 B

  const int t   = threadIdx.x;
  const int n0  = blockIdx.x * ROWS_PER_BLK;       // 128 consecutive rows, same b
  const int b   = n0 >> 12;
  const int hw0 = n0 & (NHW - 1);

  // ---- codebook + norms: contiguous async global->LDS DMA (ASYNCcnt) ----
  #pragma unroll 8
  for (int i = 0; i < 32; ++i) {
    int off = (t + 256 * i) * 8;                   // f16 elems, 16B per op
    vq_async_cp16(wsE + off, ldsE + off);
  }
  vq_async_cp16(wsEnorm + t * 4, enS + t * 4);     // 256 * 16B = 4KB

  // ---- x tile (BCHW -> [row][c] f16) on VALU, overlapping the DMA ----
  #pragma unroll
  for (int i = 0; i < 8; ++i) {
    int f4 = (t + 256 * i) * 4;                    // 4 consecutive rows, same c
    int c = f4 >> 7, r = f4 & (ROWS_PER_BLK - 1);
    float4 v = *(const float4*)(x + (((b * NC) + c) << 12) + hw0 + r);
    ldsX[(r + 0) * NC + c] = (_Float16)v.x;
    ldsX[(r + 1) * NC + c] = (_Float16)v.y;
    ldsX[(r + 2) * NC + c] = (_Float16)v.z;
    ldsX[(r + 3) * NC + c] = (_Float16)v.w;
  }
  vq_async_wait0();
  __syncthreads();

  const int lane = t & 31;
  const int wv   = t >> 5;          // 8 waves, 16 rows each
  const int rowb = wv * 16;
  const int m    = lane & 15;       // row-in-tile (A) / col-in-tile (B,C)
  const int h    = lane >> 4;       // half-wave K interleave

  // ---- A fragments (16x32 f16, per ISA layout), built once per wave ----
  v16h a[2];
  #pragma unroll
  for (int q = 0; q < 2; ++q) {
    const _Float16* ap = ldsX + (rowb + m) * NC + 32 * q + 8 * h;
    v8h lo = *(const v8h*)(ap);          // K = 32q + 8h .. +7
    v8h hi = *(const v8h*)(ap + 16);     // K = 32q + 16 + 8h .. +7
    #pragma unroll
    for (int e = 0; e < 8; ++e) { a[q][e] = lo[e]; a[q][8 + e] = hi[e]; }
  }

  float minv[8];
  int   mini[8];
  #pragma unroll
  for (int s = 0; s < 8; ++s) { minv[s] = 3.402823e38f; mini[s] = 0; }

  // ---- K-loop: 64 column tiles of 16 codes, all data from LDS.
  //      Unroll 2 -> two independent WMMA/argmin chains to hide the
  //      f16-WMMA->VALU hazard NOPs and ds_load waits. ----
  #pragma unroll 2
  for (int ct = 0; ct < 64; ++ct) {
    const _Float16* bp = ldsE + (ct * 16 + m) * NC + 8 * h;
    v8h x0 = *(const v8h*)(bp);          // c = 8h .. +7
    v8h x1 = *(const v8h*)(bp + 16);     // c = 16+8h .. +7
    v8h x2 = *(const v8h*)(bp + 32);     // c = 32+8h .. +7
    v8h x3 = *(const v8h*)(bp + 48);
    v16h b0, b1;
    #pragma unroll
    for (int e = 0; e < 8; ++e) {
      b0[e] = x0[e]; b0[8 + e] = x1[e];
      b1[e] = x2[e]; b1[8 + e] = x3[e];
    }
    v8f acc = {};
    acc = __builtin_amdgcn_wmma_f32_16x16x32_f16(false, a[0], false, b0,
                                                 (short)0, acc, false, false);
    acc = __builtin_amdgcn_wmma_f32_16x16x32_f16(false, a[1], false, b1,
                                                 (short)0, acc, false, false);
    float en = enS[ct * 16 + m];
    #pragma unroll
    for (int s = 0; s < 8; ++s) {        // slot s -> row rowb + s + 8h, col ct*16+m
      float d = en - 2.0f * acc[s];
      if (d < minv[s]) { minv[s] = d; mini[s] = ct * 16 + m; }
    }
  }

  // ---- argmin across the 16 lanes of each half-wave (first-index ties) ----
  #pragma unroll
  for (int s = 0; s < 8; ++s) {
    float v = minv[s]; int id = mini[s];
    #pragma unroll
    for (int off = 8; off >= 1; off >>= 1) {
      float ov = __shfl_xor(v, off, 32);
      int   oi = __shfl_xor(id, off, 32);
      if (ov < v || (ov == v && oi < id)) { v = ov; id = oi; }
    }
    if (m == 0) ldsI[rowb + s + 8 * h] = id;
  }
  __syncthreads();

  // ---- outputs: gather f32 codes (coalesced over hw) + fused loss ----
  float lsum = 0.f;
  #pragma unroll 4
  for (int i = 0; i < 32; ++i) {
    int f = t + 256 * i;
    int c = f >> 7, r = f & (ROWS_PER_BLK - 1);
    int k = ldsI[r];
    float qv = embed[c * NK + k];
    int ga = (((b * NC) + c) << 12) + hw0 + r;
    float xd = x[ga] - qv;
    lsum += xd * xd;
    outQ[ga] = qv;
  }
  if (t < ROWS_PER_BLK) outIdx[n0 + t] = ldsI[t];

  #pragma unroll
  for (int off = 16; off >= 1; off >>= 1) lsum += __shfl_xor(lsum, off, 32);
  if (lane == 0)
    atomicAdd(lossPtr, lsum * (1.25f / (float)(NN * NC)));  // e_loss + 0.25*q_loss
}

// ---------------------------------------------------------------------------
extern "C" void kernel_launch(void* const* d_in, const int* in_sizes, int n_in,
                              void* d_out, int out_size, void* d_ws, size_t ws_size,
                              hipStream_t stream) {
  const float* x     = (const float*)d_in[0];   // (16,64,64,64) f32
  const float* embed = (const float*)d_in[1];   // (64,1024) f32

  float* out     = (float*)d_out;
  float* outQ    = out;                          // 4194304 f32
  int*   outIdx  = (int*)(out + (size_t)NN * NC);// 65536 i32
  float* lossPtr = out + (size_t)NN * NC + NN;   // 1 f32

  _Float16* wsE     = (_Float16*)d_ws;                       // 131072 B, [k][c]
  float*    wsEnorm = (float*)((char*)d_ws + 131072);        //   4096 B

  vq_prep<<<64, 256, 0, stream>>>(embed, wsE, wsEnorm, lossPtr);

  const size_t smemBytes = 152064;   // 128K codebook + 4K norms + 16K x-tile + idx
  (void)hipFuncSetAttribute(reinterpret_cast<const void*>(vq_main),
                            hipFuncAttributeMaxDynamicSharedMemorySize,
                            (int)smemBytes);
  vq_main<<<NN / ROWS_PER_BLK, 256, smemBytes, stream>>>(
      x, embed, wsE, wsEnorm, outQ, outIdx, lossPtr);
}
